// OrientedRCNNHead_42700564857182
// MI455X (gfx1250) — matrix operand order repaired and checked by reading, not compiled, not run on hardware
//
#include <hip/hip_runtime.h>
#include <hip/hip_bf16.h>

// ---------------------------------------------------------------------------
// Types for WMMA fragments (CDNA5 / gfx1250, wave32)
// ---------------------------------------------------------------------------
typedef __attribute__((ext_vector_type(16))) __bf16 bf16x16;
typedef __attribute__((ext_vector_type(8)))  __bf16 bf16x8;
typedef __attribute__((ext_vector_type(8)))  float  fx8;

__device__ __forceinline__ unsigned short f2bf(float f) {
    unsigned int u = __float_as_uint(f);
    unsigned int r = u + 0x7FFFu + ((u >> 16) & 1u);   // round-to-nearest-even
    return (unsigned short)(r >> 16);
}

// ---------------------------------------------------------------------------
// Kernel 1: tiled transpose + f32->bf16.  W (K x N, f32) -> Wt (N x K, bf16).
// Coalesced on both global sides via a 32x32 LDS tile.
// ---------------------------------------------------------------------------
__global__ __launch_bounds__(256) void transpose_f32_to_bf16_kernel(
    const float* __restrict__ W, unsigned short* __restrict__ Wt, int K, int N) {
    __shared__ float tile[32][33];
    int k0 = blockIdx.x * 32;
    int n0 = blockIdx.y * 32;
    int c  = threadIdx.x & 31;
    int r8 = threadIdx.x >> 5;   // 0..7
    #pragma unroll
    for (int i = 0; i < 4; ++i) {
        int r = r8 + i * 8;
        tile[r][c] = W[(size_t)(k0 + r) * N + n0 + c];
    }
    __syncthreads();
    #pragma unroll
    for (int i = 0; i < 4; ++i) {
        int nn = r8 + i * 8;
        Wt[(size_t)(n0 + nn) * K + k0 + c] = f2bf(tile[c][nn]);
    }
}

// ---------------------------------------------------------------------------
// Kernel 2: rotated ROI align.  One block per roi (2048 blocks), 256 threads =
// one channel per thread.  Sample coordinates shared via LDS.
// Output: x[m, c*49 + bin] as bf16, m = b*1024 + l*256 + n, row length 12544.
// ---------------------------------------------------------------------------
__global__ __launch_bounds__(256) void roi_align_kernel(
    const float* __restrict__ f0, const float* __restrict__ f1,
    const float* __restrict__ f2, const float* __restrict__ f3,
    const float* __restrict__ proposals, unsigned short* __restrict__ xbf) {
    int bid = blockIdx.x;             // 0..2047
    int b   = bid >> 10;
    int rem = bid & 1023;
    int l   = rem >> 8;
    int n   = rem & 255;
    int HW  = 256 >> l;
    const float* feat = (l == 0) ? f0 : (l == 1) ? f1 : (l == 2) ? f2 : f3;

    __shared__ int   sy0[196], sx0[196], sy1[196], sx1[196];
    __shared__ float sly[196], slx[196];
    __shared__ unsigned char sval[196];

    int t = threadIdx.x;
    if (t < 196) {
        const float* v = proposals + (((size_t)l * 2 + b) * 256 + n) * 8;
        float cx = 0.25f * (v[0] + v[2] + v[4] + v[6]);
        float cy = 0.25f * (v[1] + v[3] + v[5] + v[7]);
        float e1x = v[2] - v[0], e1y = v[3] - v[1];
        float e2x = v[4] - v[2], e2y = v[5] - v[3];
        float w  = sqrtf(e1x * e1x + e1y * e1y);
        float h  = sqrtf(e2x * e2x + e2y * e2y);
        float th = atan2f(e1y, e1x);
        float c = cosf(th), s = sinf(th);
        int i = t / 14, j = t % 14;
        float gy = ((i + 0.5f) * (1.0f / 14.0f) - 0.5f) * h;
        float gx = ((j + 0.5f) * (1.0f / 14.0f) - 0.5f) * w;
        float px = cx + gx * c - gy * s;
        float py = cy + gx * s + gy * c;
        float Hf = (float)HW;
        bool valid = (py > -1.f) && (py < Hf) && (px > -1.f) && (px < Hf);
        float y = fminf(fmaxf(py, 0.f), Hf - 1.f);
        float x = fminf(fmaxf(px, 0.f), Hf - 1.f);
        int y0 = (int)floorf(y);
        int x0 = (int)floorf(x);
        int y1 = min(y0 + 1, HW - 1);
        int x1 = min(x0 + 1, HW - 1);
        sy0[t] = y0; sx0[t] = x0; sy1[t] = y1; sx1[t] = x1;
        sly[t] = y - (float)y0; slx[t] = x - (float)x0;
        sval[t] = valid ? 1 : 0;
    }
    __syncthreads();

    int c = t;  // channel 0..255
    const float* fm = feat + ((size_t)(b * 256 + c)) * HW * HW;
    unsigned short* orow = xbf + (size_t)bid * 12544 + (size_t)c * 49;
    #pragma unroll 1
    for (int bin = 0; bin < 49; ++bin) {
        int bi = bin / 7, bj = bin % 7;
        float sum = 0.f;
        #pragma unroll
        for (int di = 0; di < 2; ++di)
        #pragma unroll
        for (int dj = 0; dj < 2; ++dj) {
            int p = (2 * bi + di) * 14 + (2 * bj + dj);
            if (sval[p]) {
                int y0 = sy0[p], x0 = sx0[p], y1 = sy1[p], x1 = sx1[p];
                float ly = sly[p], lx = slx[p];
                float hy = 1.f - ly, hx = 1.f - lx;
                float v00 = fm[y0 * HW + x0];
                float v01 = fm[y0 * HW + x1];
                float v10 = fm[y1 * HW + x0];
                float v11 = fm[y1 * HW + x1];
                sum += hy * hx * v00 + hy * lx * v01 + ly * hx * v10 + ly * lx * v11;
            }
        }
        orow[bin] = f2bf(sum * 0.25f);
    }
}

// ---------------------------------------------------------------------------
// Kernel 3: bf16 WMMA GEMM with async-to-LDS double buffering.
//   C = relu(A(MxK) * Bt(NxK)^T + bias)
// Block tile 64x128, 8 waves (4 M-subtiles x 2 N-halves), 4 accumulators/wave.
// Both A and Bt are row-contiguous in K, so every global->LDS chunk is one
// GLOBAL_LOAD_ASYNC_TO_LDS_B128 (ASYNCcnt), double-buffered with
// s_wait_asynccnt 3 (async loads complete in order).
// ---------------------------------------------------------------------------
#define BM   64
#define BN   128
#define BK   32
#define LPAD 48   // padded LDS row length in ushorts (96B, 16B-aligned)

template<bool BF16OUT>
__global__ __launch_bounds__(256) void gemm_bf16_kernel(
    const unsigned short* __restrict__ A,    // M x K row-major (bf16 bits)
    const unsigned short* __restrict__ Bt,   // N x K row-major (bf16 bits)
    const float* __restrict__ bias,          // [N]
    void* __restrict__ outP,                 // M x N (bf16 or f32)
    int M, int N, int K, int relu) {
    __shared__ __align__(16) unsigned short As[2][BM * LPAD];
    __shared__ __align__(16) unsigned short Bs[2][BN * LPAD];

    int tid   = threadIdx.x;
    int lane  = tid & 31;
    int wave  = tid >> 5;
    int wm    = wave & 3;    // M subtile (16 rows each)
    int wn    = wave >> 2;   // N half (64 cols each)
    int blockM = blockIdx.y * BM;
    int blockN = blockIdx.x * BN;
    int lhalf = lane >> 4;   // 0 or 1
    int l15   = lane & 15;

    fx8 acc[4];
    #pragma unroll
    for (int s = 0; s < 4; ++s)
        #pragma unroll
        for (int e = 0; e < 8; ++e) acc[s][e] = 0.f;

    // Per-thread 16B chunk mapping (rows x 32-k tile, 4 chunks of 8 ushorts).
    int ar = tid >> 2;            // A row 0..63 / B rows 0..63 (+64 for 2nd)
    int ac = (tid & 3) * 8;       // k segment 0,8,16,24
    const unsigned short* Ag  = A  + (size_t)(blockM + ar) * K + ac;
    const unsigned short* Bg0 = Bt + (size_t)(blockN + ar) * K + ac;
    const unsigned short* Bg1 = Bt + (size_t)(blockN + 64 + ar) * K + ac;

    unsigned laA[2], laB0[2], laB1[2];
    #pragma unroll
    for (int bufi = 0; bufi < 2; ++bufi) {
        laA[bufi]  = (unsigned)(uintptr_t)&As[bufi][ar * LPAD + ac];
        laB0[bufi] = (unsigned)(uintptr_t)&Bs[bufi][ar * LPAD + ac];
        laB1[bufi] = (unsigned)(uintptr_t)&Bs[bufi][(64 + ar) * LPAD + ac];
    }

    auto issue_tile = [&](int bufi, int k0) {
        unsigned long long ga  = (unsigned long long)(uintptr_t)(Ag + k0);
        unsigned long long gb0 = (unsigned long long)(uintptr_t)(Bg0 + k0);
        unsigned long long gb1 = (unsigned long long)(uintptr_t)(Bg1 + k0);
        asm volatile("global_load_async_to_lds_b128 %0, %1, off"
                     :: "v"(laA[bufi]), "v"(ga) : "memory");
        asm volatile("global_load_async_to_lds_b128 %0, %1, off"
                     :: "v"(laB0[bufi]), "v"(gb0) : "memory");
        asm volatile("global_load_async_to_lds_b128 %0, %1, off"
                     :: "v"(laB1[bufi]), "v"(gb1) : "memory");
    };

    int nk = K / BK;
    issue_tile(0, 0);

    for (int kk = 0; kk < nk; ++kk) {
        int buf = kk & 1;
        if (kk + 1 < nk) {
            issue_tile(buf ^ 1, (kk + 1) * BK);
            // 3 newer async instrs outstanding; <=3 means current tile landed.
            asm volatile("s_wait_asynccnt 3" ::: "memory");
        } else {
            asm volatile("s_wait_asynccnt 0" ::: "memory");
        }
        __syncthreads();

        // A fragment: lane<16 -> M=l15, K {0..7,16..23}; lane>=16 -> K {8..15,24..31}
        int am = wm * 16 + l15;
        const bf16x8* arow = (const bf16x8*)&As[buf][am * LPAD];
        union { bf16x16 v; bf16x8 h[2]; } ua;
        ua.h[0] = arow[lhalf];
        ua.h[1] = arow[2 + lhalf];
        bf16x16 afrag = ua.v;

        #pragma unroll
        for (int sub = 0; sub < 4; ++sub) {
            // B fragment: lane<16 -> N=l15, K {0..7,16..23}; lane>=16 -> K {8..15,24..31}
            int bcol = wn * 64 + sub * 16 + l15;
            const bf16x8* brow = (const bf16x8*)&Bs[buf][bcol * LPAD];
            union { bf16x16 v; bf16x8 h[2]; } ub;
            ub.h[0] = brow[lhalf];
            ub.h[1] = brow[2 + lhalf];
            acc[sub] = __builtin_amdgcn_wmma_f32_16x16x32_bf16(
                false, afrag, false, ub.v, (short)0, acc[sub], false, false);
        }
        __syncthreads();
    }

    // Epilogue: D layout — vgpr i holds M = i + 8*lhalf, lane%16 = N.
    #pragma unroll
    for (int sub = 0; sub < 4; ++sub) {
        int nCol = blockN + wn * 64 + sub * 16 + l15;
        float bv = bias[nCol];
        #pragma unroll
        for (int i = 0; i < 8; ++i) {
            int mRow = blockM + wm * 16 + lhalf * 8 + i;
            float v = acc[sub][i] + bv;
            if (relu) v = fmaxf(v, 0.f);
            if (BF16OUT) ((unsigned short*)outP)[(size_t)mRow * N + nCol] = f2bf(v);
            else         ((float*)outP)[(size_t)mRow * N + nCol] = v;
        }
    }
}

// ---------------------------------------------------------------------------
// Kernel 4: heads (cls 11 + reg 5), softmax, rotated-box decode.
// One wave per output row m; lanes 0..15 each compute one dot product.
// ---------------------------------------------------------------------------
__global__ __launch_bounds__(256) void head_kernel(
    const float* __restrict__ h2, const float* __restrict__ Wc,
    const float* __restrict__ bc, const float* __restrict__ Wr,
    const float* __restrict__ br, const float* __restrict__ proposals,
    float* __restrict__ out) {
    int wave = threadIdx.x >> 5;
    int lane = threadIdx.x & 31;
    int m = blockIdx.x * 8 + wave;   // 0..2047

    int j = lane & 15;
    const float* Wp; int ldw; float bb;
    if (j < 11) { Wp = Wc + j;        ldw = 11; bb = bc[j]; }
    else        { Wp = Wr + (j - 11); ldw = 5;  bb = br[j - 11]; }

    const float* xr = h2 + (size_t)m * 1024;
    float s = 0.f;
    #pragma unroll 4
    for (int k = 0; k < 1024; ++k) s += xr[k] * Wp[(size_t)k * ldw];
    s += bb;

    float sv[16];
    #pragma unroll
    for (int q = 0; q < 16; ++q) sv[q] = __shfl(s, q, 32);

    if (lane == 0) {
        // softmax over 11 classes
        float mx = sv[0];
        #pragma unroll
        for (int q = 1; q < 11; ++q) mx = fmaxf(mx, sv[q]);
        float ex[11]; float ssum = 0.f;
        #pragma unroll
        for (int q = 0; q < 11; ++q) { ex[q] = expf(sv[q] - mx); ssum += ex[q]; }
        float inv = 1.f / ssum;

        // proposal -> (cx,cy,w,h,theta)
        int b = m >> 10; int rem = m & 1023; int l = rem >> 8; int n = rem & 255;
        const float* v = proposals + (((size_t)l * 2 + b) * 256 + n) * 8;
        float cx = 0.25f * (v[0] + v[2] + v[4] + v[6]);
        float cy = 0.25f * (v[1] + v[3] + v[5] + v[7]);
        float e1x = v[2] - v[0], e1y = v[3] - v[1];
        float e2x = v[4] - v[2], e2y = v[5] - v[3];
        float w  = sqrtf(e1x * e1x + e1y * e1y);
        float h  = sqrtf(e2x * e2x + e2y * e2y);
        float th = atan2f(e1y, e1x);
        const float kStrides[4] = {4.f, 8.f, 16.f, 32.f};
        float stride = kStrides[l];

        const float CLAMP_V = 4.1351665567423557f;   // |log(16/1000)|
        float r0 = sv[11], r1 = sv[12];
        float dw = fminf(fmaxf(sv[13], -CLAMP_V), CLAMP_V);
        float dh = fminf(fmaxf(sv[14], -CLAMP_V), CLAMP_V);
        float bx = w * r0 + cx;
        float by = h * r1 + cy;
        float bw = w * expf(dw);
        float bh = h * expf(dh);
        float ba = th + sv[15];

        float* o = out + (size_t)m * 16;
        #pragma unroll
        for (int q = 0; q < 11; ++q) o[q] = ex[q] * inv;
        o[11] = bx * stride;
        o[12] = by * stride;
        o[13] = bw * stride;
        o[14] = bh * stride;
        o[15] = ba;
    }
}

// ---------------------------------------------------------------------------
// Launch
// ---------------------------------------------------------------------------
extern "C" void kernel_launch(void* const* d_in, const int* in_sizes, int n_in,
                              void* d_out, int out_size, void* d_ws, size_t ws_size,
                              hipStream_t stream) {
    (void)in_sizes; (void)n_in; (void)out_size; (void)ws_size;
    const float* f0        = (const float*)d_in[0];
    const float* f1        = (const float*)d_in[1];
    const float* f2        = (const float*)d_in[2];
    const float* f3        = (const float*)d_in[3];
    const float* proposals = (const float*)d_in[4];
    const float* W1        = (const float*)d_in[5];
    const float* b1        = (const float*)d_in[6];
    const float* W2        = (const float*)d_in[7];
    const float* b2        = (const float*)d_in[8];
    const float* Wc        = (const float*)d_in[9];
    const float* bc        = (const float*)d_in[10];
    const float* Wr        = (const float*)d_in[11];
    const float* br        = (const float*)d_in[12];
    float* out = (float*)d_out;

    char* ws = (char*)d_ws;
    size_t off = 0;
    auto wsAlloc = [&](size_t bytes) -> char* {
        char* p = ws + off;
        off += (bytes + 255) & ~(size_t)255;
        return p;
    };
    unsigned short* xbf = (unsigned short*)wsAlloc((size_t)2048 * 12544 * 2);
    unsigned short* w1t = (unsigned short*)wsAlloc((size_t)12544 * 1024 * 2);
    unsigned short* w2t = (unsigned short*)wsAlloc((size_t)1024 * 1024 * 2);
    unsigned short* h1bf = (unsigned short*)wsAlloc((size_t)2048 * 1024 * 2);
    float*          h2   = (float*)wsAlloc((size_t)2048 * 1024 * 4);

    // Transpose + convert weights: W1 (12544x1024) -> w1t (1024x12544),
    // W2 (1024x1024) -> w2t (1024x1024).
    transpose_f32_to_bf16_kernel<<<dim3(12544 / 32, 1024 / 32), 256, 0, stream>>>(
        W1, w1t, 12544, 1024);
    transpose_f32_to_bf16_kernel<<<dim3(1024 / 32, 1024 / 32), 256, 0, stream>>>(
        W2, w2t, 1024, 1024);

    roi_align_kernel<<<2048, 256, 0, stream>>>(f0, f1, f2, f3, proposals, xbf);

    // GEMM1: x(2048x12544) @ W1 + b1, relu -> h1 (bf16)
    gemm_bf16_kernel<true><<<dim3(1024 / BN, 2048 / BM), 256, 0, stream>>>(
        xbf, w1t, b1, (void*)h1bf, 2048, 1024, 12544, 1);

    // GEMM2: h1(2048x1024) @ W2 + b2, relu -> h2 (f32)
    gemm_bf16_kernel<false><<<dim3(1024 / BN, 2048 / BM), 256, 0, stream>>>(
        h1bf, w2t, b2, (void*)h2, 2048, 1024, 1024, 1);

    head_kernel<<<256, 256, 0, stream>>>(h2, Wc, bc, Wr, br, proposals, out);
}
